// GATPredictor_84954453115321
// MI455X (gfx1250) — matrix operand order repaired and checked by reading, not compiled, not run on hardware
//
#include <hip/hip_runtime.h>
#include <hip/hip_bf16.h>

#define BDIM 8
#define TDIM 12
#define NDIM 1500
#define HID 64
#define HEADS 4
#define DH 16
#define OUTD 3
#define BN (BDIM * NDIM)   // 12000

typedef __attribute__((ext_vector_type(16))) _Float16 v16h;
typedef __attribute__((ext_vector_type(8)))  float    v8f;
typedef __attribute__((ext_vector_type(4)))  unsigned int u32x4;
typedef __attribute__((ext_vector_type(8)))  int      i32x8;
typedef __attribute__((ext_vector_type(4)))  int      i32x4;

// ---------------------------------------------------------------------------
// WMMA helpers (layouts per CDNA5 ISA §7.12.2, wave32)
// ---------------------------------------------------------------------------
__device__ __forceinline__ v8f wmma_f16(v16h a, v16h b, v8f c) {
  return __builtin_amdgcn_wmma_f32_16x16x32_f16(
      /*neg_a=*/false, a, /*neg_b=*/false, b,
      /*c_mod=*/(short)0, c, /*reuse_a=*/false, /*reuse_b=*/false);
}

// A fragment: 16x32 f16, element (m,k) = src[m*ld + k]
__device__ __forceinline__ v16h load_a_frag(const _Float16* src, int ld) {
  int l = threadIdx.x & 31;
  int m  = l & 15;
  int kb = (l < 16) ? 0 : 8;
  v16h a;
#pragma unroll
  for (int v = 0; v < 8; ++v) {
    int k = kb + ((v & 3) * 2) + ((v >> 2) * 16);
    a[2 * v]     = src[m * ld + k];
    a[2 * v + 1] = src[m * ld + k + 1];
  }
  return a;
}

// B fragment: 32x16 f16, element (k,n) = src[n*ld + k]   (B supplied transposed,
// i.e. weight-matrix row-major W[gate][k])
__device__ __forceinline__ v16h load_b_fragT(const _Float16* src, int ld) {
  int l = threadIdx.x & 31;
  int n  = l & 15;
  int kb = (l < 16) ? 0 : 8;
  v16h b;
#pragma unroll
  for (int v = 0; v < 8; ++v) {
    int k = kb + ((v & 3) * 2) + ((v >> 2) * 16);
    b[2 * v]     = src[n * ld + k];
    b[2 * v + 1] = src[n * ld + k + 1];
  }
  return b;
}

// B fragment: 32x16 f16, element (k,n) = src[k*ld + n]   (row-major K x N)
__device__ __forceinline__ v16h load_b_frag(const _Float16* src, int ld) {
  int l = threadIdx.x & 31;
  int n  = l & 15;
  int kb = (l < 16) ? 0 : 8;
  v16h b;
#pragma unroll
  for (int v = 0; v < 8; ++v) {
    int k = kb + ((v & 3) * 2) + ((v >> 2) * 16);
    b[2 * v]     = src[k * ld + n];
    b[2 * v + 1] = src[(k + 1) * ld + n];
  }
  return b;
}

__device__ __forceinline__ float sigf(float x) { return 1.f / (1.f + __expf(-x)); }

// ---------------------------------------------------------------------------
// Tensor Data Mover: DMA a 2D f16 tile (tile_w elems x tile_h rows, row stride
// row_stride elems) from global memory into LDS. Descriptor layout per
// CDNA5 ISA (D# groups 0/1); groups 2/3 unused (2D).
// Tracked by TENSORcnt.
// ---------------------------------------------------------------------------
__device__ __forceinline__ void tdm_load_tile_f16(
    unsigned lds_byte_off, const void* gaddr,
    unsigned tile_w, unsigned tile_h, unsigned row_stride)
{
  unsigned long long ga = (unsigned long long)(uintptr_t)gaddr;
  u32x4 g0;
  g0[0] = 1u;                                          // count=1 valid user D#
  g0[1] = lds_byte_off;                                // lds_addr (bytes)
  g0[2] = (unsigned)(ga & 0xFFFFFFFFu);                // global_addr[31:0]
  g0[3] = (unsigned)((ga >> 32) & 0x01FFFFFFu)         // global_addr[56:32]
          | (2u << 30);                                // type=2 ("image")
  i32x8 g1;
  g1[0] = (int)(1u << 16);                             // wg_mask=0, data_size=1 (2B)
  g1[1] = (int)((row_stride & 0xFFFFu) << 16);         // tensor_dim0[15:0]
  g1[2] = (int)(((row_stride >> 16) & 0xFFFFu)         // tensor_dim0[31:16]
          | ((tile_h & 0xFFFFu) << 16));               // tensor_dim1[15:0]
  g1[3] = (int)(((tile_h >> 16) & 0xFFFFu)             // tensor_dim1[31:16]
          | ((tile_w & 0xFFFFu) << 16));               // tile_dim0
  g1[4] = (int)(tile_h & 0xFFFFu);                     // tile_dim1 (tile_dim2=0)
  g1[5] = (int)row_stride;                             // tensor_dim0_stride[31:0]
  g1[6] = 0;                                           // stride0[47:32], stride1 lo
  g1[7] = 0;
  i32x4 z4 = {0, 0, 0, 0};
#if defined(__clang_major__) && (__clang_major__ >= 23)
  i32x8 z8 = {0, 0, 0, 0, 0, 0, 0, 0};
  __builtin_amdgcn_tensor_load_to_lds(g0, g1, z4, z4, z8, 0);
#else
  __builtin_amdgcn_tensor_load_to_lds(g0, g1, z4, z4, 0);
#endif
}

__device__ __forceinline__ unsigned lds_offset_of(const void* p) {
  // generic shared address = {shared aperture, 32-bit LDS offset}
  return (unsigned)(uintptr_t)p;
}

// ---------------------------------------------------------------------------
// Kernel 1: fused 2-layer LSTM over T=12 steps. 2 waves/block, 16 seqs/wave.
// Weights (3 x 256x64) resident in LDS as f16; h/c states resident in LDS.
// Recurrent GEMMs done with v_wmma_f32_16x16x32_f16.
// ---------------------------------------------------------------------------
__global__ __launch_bounds__(64) void lstm_kernel(
    const float* __restrict__ x,
    const float* __restrict__ Wih0, const float* __restrict__ Whh0,
    const float* __restrict__ bih0, const float* __restrict__ bhh0,
    const float* __restrict__ Wih1, const float* __restrict__ Whh1,
    const float* __restrict__ bih1, const float* __restrict__ bhh1,
    float* __restrict__ Hout)
{
  __shared__ _Float16 sW[3][256][64];      // Whh0, Wih1, Whh1 (f16)  96 KB
  __shared__ float    sBias[2][256];       // fused biases per layer
  __shared__ float    sWih0[256];          // layer-0 input column
  __shared__ float    sG[2][16][256];      // gate pre-activations per wave
  __shared__ _Float16 sH[2][2][16][64];    // h0,h1 per wave (f16, WMMA A staging)
  __shared__ float    sC[2][2][16][64];    // c0,c1 per wave
  __shared__ float    sX[2][16];

  const int tid = threadIdx.x;
  for (int i = tid; i < 256 * 64; i += blockDim.x) {
    int r = i >> 6, c = i & 63;
    sW[0][r][c] = (_Float16)Whh0[i];
    sW[1][r][c] = (_Float16)Wih1[i];
    sW[2][r][c] = (_Float16)Whh1[i];
  }
  for (int i = tid; i < 256; i += blockDim.x) {
    sBias[0][i] = bih0[i] + bhh0[i];
    sBias[1][i] = bih1[i] + bhh1[i];
    sWih0[i]    = Wih0[i];
  }

  const int wave = tid >> 5, lane = tid & 31;
  _Float16* h0 = &sH[wave][0][0][0];
  _Float16* h1 = &sH[wave][1][0][0];
  float*    c0 = &sC[wave][0][0][0];
  float*    c1 = &sC[wave][1][0][0];
  float*    G  = &sG[wave][0][0];
  for (int i = lane; i < 16 * 64; i += 32) {
    h0[i] = (_Float16)0.f; h1[i] = (_Float16)0.f;
    c0[i] = 0.f;           c1[i] = 0.f;
  }
  __syncthreads();

  const long rowbase = (long)(blockIdx.x * 2 + wave) * 16;   // 375 blocks * 32 = 12000 exact

  for (int t = 0; t < TDIM; ++t) {
    // stage x_t for the 16 sequences of this wave
    if (lane < 16) {
      long r = rowbase + lane;
      int b = (int)(r / NDIM), n = (int)(r % NDIM);
      sX[wave][lane] = x[((long)b * TDIM + t) * NDIM + n];
    }
    __syncthreads();

    // ---- layer 0: G = h0 @ Whh0^T (+ bias + x*Wih0) ----
    {
      v16h a0 = load_a_frag(h0, 64);
      v16h a1 = load_a_frag(h0 + 32, 64);
      const int nn = lane & 15, mb = (lane < 16) ? 0 : 8;
#pragma unroll 4
      for (int nt = 0; nt < 16; ++nt) {
        v8f acc = {};
        acc = wmma_f16(a0, load_b_fragT(&sW[0][nt * 16][0], 64), acc);
        acc = wmma_f16(a1, load_b_fragT(&sW[0][nt * 16][0] + 32, 64), acc);
        int gc = nt * 16 + nn;
#pragma unroll
        for (int v = 0; v < 8; ++v) {
          int m = mb + v;
          G[m * 256 + gc] = acc[v] + sBias[0][gc] + sX[wave][m] * sWih0[gc];
        }
      }
    }
    __syncthreads();
    for (int idx = lane; idx < 16 * 64; idx += 32) {
      int m = idx >> 6, j = idx & 63;
      float gi = G[m * 256 + j],       gf = G[m * 256 + 64 + j];
      float gg = G[m * 256 + 128 + j], go = G[m * 256 + 192 + j];
      float c = sigf(gf) * c0[idx] + sigf(gi) * tanhf(gg);
      c0[idx] = c;
      h0[idx] = (_Float16)(sigf(go) * tanhf(c));
    }
    __syncthreads();

    // ---- layer 1: G = h0 @ Wih1^T + h1 @ Whh1^T + bias ----
    {
      v16h p0 = load_a_frag(h0, 64);
      v16h p1 = load_a_frag(h0 + 32, 64);
      v16h q0 = load_a_frag(h1, 64);
      v16h q1 = load_a_frag(h1 + 32, 64);
      const int nn = lane & 15, mb = (lane < 16) ? 0 : 8;
#pragma unroll 4
      for (int nt = 0; nt < 16; ++nt) {
        v8f acc = {};
        acc = wmma_f16(p0, load_b_fragT(&sW[1][nt * 16][0], 64), acc);
        acc = wmma_f16(p1, load_b_fragT(&sW[1][nt * 16][0] + 32, 64), acc);
        acc = wmma_f16(q0, load_b_fragT(&sW[2][nt * 16][0], 64), acc);
        acc = wmma_f16(q1, load_b_fragT(&sW[2][nt * 16][0] + 32, 64), acc);
        int gc = nt * 16 + nn;
#pragma unroll
        for (int v = 0; v < 8; ++v) {
          int m = mb + v;
          G[m * 256 + gc] = acc[v] + sBias[1][gc];
        }
      }
    }
    __syncthreads();
    for (int idx = lane; idx < 16 * 64; idx += 32) {
      int m = idx >> 6, j = idx & 63;
      float gi = G[m * 256 + j],       gf = G[m * 256 + 64 + j];
      float gg = G[m * 256 + 128 + j], go = G[m * 256 + 192 + j];
      float c = sigf(gf) * c1[idx] + sigf(gi) * tanhf(gg);
      c1[idx] = c;
      h1[idx] = (_Float16)(sigf(go) * tanhf(c));
    }
    __syncthreads();
  }

  for (int idx = lane; idx < 16 * 64; idx += 32) {
    int m = idx >> 6, j = idx & 63;
    Hout[(rowbase + m) * 64 + j] = (float)h1[idx];
  }
}

// ---------------------------------------------------------------------------
// Kernel 2: GAT linear (hp = h @ W^T via WMMA) + attention scalars e_src/e_dst
// ---------------------------------------------------------------------------
__global__ __launch_bounds__(64) void gat_prep_kernel(
    const float* __restrict__ Hin, const float* __restrict__ W,
    const float* __restrict__ asrc, const float* __restrict__ adst,
    _Float16* __restrict__ hp, float* __restrict__ esrc, float* __restrict__ edst)
{
  __shared__ _Float16 sW[64][64];
  __shared__ _Float16 sA[2][16][64];
  __shared__ float    sHp[2][16][64];

  const int tid = threadIdx.x;
  for (int i = tid; i < 64 * 64; i += blockDim.x)
    sW[i >> 6][i & 63] = (_Float16)W[i];
  if (blockIdx.x == 0) {                       // zero the 32-row pad tail of hp
    for (int i = tid; i < 32 * 64; i += blockDim.x)
      hp[(long)BN * 64 + i] = (_Float16)0.f;
  }

  const int wave = tid >> 5, lane = tid & 31;
  const long rowbase = (long)(blockIdx.x * 2 + wave) * 16;
  for (int idx = lane; idx < 16 * 64; idx += 32) {
    int m = idx >> 6, k = idx & 63;
    sA[wave][m][k] = (_Float16)Hin[(rowbase + m) * 64 + k];
  }
  __syncthreads();

  v16h a0 = load_a_frag(&sA[wave][0][0], 64);
  v16h a1 = load_a_frag(&sA[wave][0][0] + 32, 64);
  const int nn = lane & 15, mb = (lane < 16) ? 0 : 8;
#pragma unroll
  for (int nt = 0; nt < 4; ++nt) {
    v8f acc = {};
    acc = wmma_f16(a0, load_b_fragT(&sW[nt * 16][0], 64), acc);
    acc = wmma_f16(a1, load_b_fragT(&sW[nt * 16][0] + 32, 64), acc);
    int gc = nt * 16 + nn;
#pragma unroll
    for (int v = 0; v < 8; ++v) {
      int m = mb + v;
      sHp[wave][m][gc] = acc[v];
      hp[(rowbase + m) * 64 + gc] = (_Float16)acc[v];
    }
  }
  __syncthreads();

  for (int task = lane; task < 16 * HEADS; task += 32) {
    int m = task >> 2, h = task & 3;
    float ss = 0.f, sd = 0.f;
#pragma unroll
    for (int d = 0; d < DH; ++d) {
      float v = sHp[wave][m][h * DH + d];
      ss += v * asrc[h * DH + d];
      sd += v * adst[h * DH + d];
    }
    esrc[(rowbase + m) * HEADS + h] = ss;
    edst[(rowbase + m) * HEADS + h] = sd;
  }
}

// ---------------------------------------------------------------------------
// Kernel 3: fused masked LeakyReLU-softmax + neighbor aggregation via WMMA.
// Grid: (ceil(N/16), B). Block 128 = 4 waves; wave == head.
// alpha(16 x N) @ hp(N x 16) as K=32 WMMA chunks. The hp B-tiles (32 rows x
// 16 dims f16, row stride 64) are DMA'd into LDS by the Tensor Data Mover,
// double-buffered and pipelined against the exp() A-fragment build
// (s_wait_tensorcnt 1 overlaps chunk c+1's DMA with chunk c's compute).
// ---------------------------------------------------------------------------
__global__ __launch_bounds__(128) void gat_attn_kernel(
    const _Float16* __restrict__ hp, const float* __restrict__ esrc,
    const float* __restrict__ edst, const int* __restrict__ adj,
    float* __restrict__ Hout)
{
  __shared__ float    sSum[HEADS][16];
  __shared__ _Float16 sB[HEADS][2][32][16];   // TDM destination, ping-pong

  const int i0 = blockIdx.x * 16;
  const int b  = blockIdx.y;
  const int h  = threadIdx.x >> 5;          // head = wave
  const int l  = threadIdx.x & 31;
  const int ri = l & 15;
  const int i  = min(i0 + ri, NDIM - 1);
  const float es = esrc[((long)b * NDIM + i) * HEADS + h];

  // pass 1: masked row max over j (lanes l and l+16 split even/odd j)
  float mmax = -1e30f;
  for (int j = (l >> 4); j < NDIM; j += 2) {
    if (adj[(long)i * NDIM + j] != 0) {
      float le = es + edst[((long)b * NDIM + j) * HEADS + h];
      le = le > 0.f ? le : 0.2f * le;
      mmax = fmaxf(mmax, le);
    }
  }
  mmax = fmaxf(mmax, __shfl_xor(mmax, 16));

  // pass 2: TDM-fed WMMA accumulation with on-the-fly softmax numerators
  const _Float16* hp_head = hp + (long)b * NDIM * 64 + h * DH;  // row j at +j*64
  const int NC = (NDIM + 31) / 32;            // 47 chunks (pad rows cover tail)

  // prefetch chunk 0 into buffer 0
  tdm_load_tile_f16(lds_offset_of(&sB[h][0][0][0]), hp_head, DH, 32, 64);

  v8f acc = {};
  float rsum = 0.f;
  const int kb = (l < 16) ? 0 : 8;
  for (int c = 0; c < NC; ++c) {
    const int j0 = c * 32;
    if (c + 1 < NC) {
      tdm_load_tile_f16(lds_offset_of(&sB[h][(c + 1) & 1][0][0]),
                        hp_head + (long)(j0 + 32) * 64, DH, 32, 64);
      __builtin_amdgcn_s_wait_tensorcnt(1);   // oldest (chunk c) complete
    } else {
      __builtin_amdgcn_s_wait_tensorcnt(0);
    }

    v16h a;
#pragma unroll
    for (int v = 0; v < 8; ++v) {
      int k = kb + ((v & 3) * 2) + ((v >> 2) * 16);
#pragma unroll
      for (int u = 0; u < 2; ++u) {
        int j = j0 + k + u;
        float ev = 0.f;
        if (j < NDIM && adj[(long)i * NDIM + j] != 0) {
          float le = es + edst[((long)b * NDIM + j) * HEADS + h];
          le = le > 0.f ? le : 0.2f * le;
          ev = __expf(le - mmax);
        }
        a[2 * v + u] = (_Float16)ev;
        rsum += ev;
      }
    }
    v16h bf = load_b_frag(&sB[h][c & 1][0][0], DH);
    acc = wmma_f16(a, bf, acc);
  }
  rsum += __shfl_xor(rsum, 16);
  if (l < 16) sSum[h][l] = rsum;
  __syncthreads();

  const int n = l & 15, mb2 = (l < 16) ? 0 : 8;
#pragma unroll
  for (int v = 0; v < 8; ++v) {
    int m = mb2 + v, iout = i0 + m;
    if (iout < NDIM) {
      float s = sSum[h][m];
      float val = (s > 0.f) ? acc[v] / s : 0.f;
      Hout[((long)b * NDIM + iout) * 64 + h * DH + n] = fmaxf(val, 0.f);  // ReLU
    }
  }
}

// ---------------------------------------------------------------------------
// Kernel 4: output head  out[b,o,n] = H[b*N+n,:] . out_W[o,:] + out_b[o]
// ---------------------------------------------------------------------------
__global__ __launch_bounds__(256) void out_proj_kernel(
    const float* __restrict__ H, const float* __restrict__ W,
    const float* __restrict__ bv, float* __restrict__ out)
{
  int r = blockIdx.x * blockDim.x + threadIdx.x;
  if (r >= BN) return;
  int b = r / NDIM, n = r % NDIM;
#pragma unroll
  for (int o = 0; o < OUTD; ++o) {
    float s = bv[o];
#pragma unroll
    for (int k = 0; k < HID; ++k) s += H[(long)r * HID + k] * W[o * HID + k];
    out[((long)b * OUTD + o) * NDIM + n] = s;
  }
}

// ---------------------------------------------------------------------------
extern "C" void kernel_launch(void* const* d_in, const int* in_sizes, int n_in,
                              void* d_out, int out_size, void* d_ws, size_t ws_size,
                              hipStream_t stream) {
  const float* x         = (const float*)d_in[0];
  const int*   adj       = (const int*)  d_in[1];
  const float* lstm_Wih0 = (const float*)d_in[2];
  const float* lstm_Whh0 = (const float*)d_in[3];
  const float* lstm_bih0 = (const float*)d_in[4];
  const float* lstm_bhh0 = (const float*)d_in[5];
  const float* lstm_Wih1 = (const float*)d_in[6];
  const float* lstm_Whh1 = (const float*)d_in[7];
  const float* lstm_bih1 = (const float*)d_in[8];
  const float* lstm_bhh1 = (const float*)d_in[9];
  const float* gat0_W    = (const float*)d_in[10];
  const float* gat0_asrc = (const float*)d_in[11];
  const float* gat0_adst = (const float*)d_in[12];
  const float* gat1_W    = (const float*)d_in[13];
  const float* gat1_asrc = (const float*)d_in[14];
  const float* gat1_adst = (const float*)d_in[15];
  const float* out_W     = (const float*)d_in[16];
  const float* out_b     = (const float*)d_in[17];

  // workspace layout
  char* ws = (char*)d_ws;
  float*    H0 = (float*)ws;                               // 12000*64 f32
  float*    H1 = (float*)(ws + 3072000);                   // 12000*64 f32
  _Float16* hp = (_Float16*)(ws + 6144000);                // (12000+32)*64 f16
  float*    es = (float*)(ws + 6144000 + 1540096);         // 12000*4  f32
  float*    ed = (float*)(ws + 6144000 + 1540096 + 192000);// 12000*4  f32

  const dim3 lstmGrid(BN / 32);        // 375, exact
  const dim3 prepGrid(BN / 32);        // 375, exact
  const dim3 attnGrid((NDIM + 15) / 16, BDIM);   // (94, 8)

  // 1) fused 2-layer LSTM -> H0 (last-step hidden of top layer)
  lstm_kernel<<<lstmGrid, 64, 0, stream>>>(
      x, lstm_Wih0, lstm_Whh0, lstm_bih0, lstm_bhh0,
      lstm_Wih1, lstm_Whh1, lstm_bih1, lstm_bhh1, H0);

  // 2) GAT layer 0
  gat_prep_kernel<<<prepGrid, 64, 0, stream>>>(H0, gat0_W, gat0_asrc, gat0_adst, hp, es, ed);
  gat_attn_kernel<<<attnGrid, 128, 0, stream>>>(hp, es, ed, adj, H1);

  // 3) GAT layer 1
  gat_prep_kernel<<<prepGrid, 64, 0, stream>>>(H1, gat1_W, gat1_asrc, gat1_adst, hp, es, ed);
  gat_attn_kernel<<<attnGrid, 128, 0, stream>>>(hp, es, ed, adj, H0);

  // 4) output head -> (B, OUT, N)
  out_proj_kernel<<<dim3((BN + 255) / 256), 256, 0, stream>>>(H0, out_W, out_b, (float*)d_out);
}
